// HeteroData_GNNmodel_9294309228905
// MI455X (gfx1250) — matrix operand order repaired and checked by reading, not compiled, not run on hardware
//
#include <hip/hip_runtime.h>

// ---------------------------------------------------------------------------
// Problem constants (match reference)
// ---------------------------------------------------------------------------
#define NG   30000
#define NC   5000
#define FIN  256
#define HH1  256
#define HH2  128
#define EGG  480000
#define ECC  160000
#define ELBL 100000

typedef __attribute__((ext_vector_type(16))) __bf16 v16bf;
typedef __attribute__((ext_vector_type(8)))  float  v8f;

struct alignas(16) U4 { unsigned x, y, z, w; };

__device__ __forceinline__ unsigned pack_bf16(float lo, float hi) {
    unsigned a = __float_as_uint(lo), b = __float_as_uint(hi);
    a = (a + 0x7FFFu + ((a >> 16) & 1u)) >> 16;   // round-to-nearest-even
    b = (b + 0x7FFFu + ((b >> 16) & 1u)) >> 16;
    return (b << 16) | (a & 0xFFFFu);
}

// ---------------------------------------------------------------------------
// Utility kernels
// ---------------------------------------------------------------------------
__global__ void zero_f32(float* p, int n) {
    int i = blockIdx.x * blockDim.x + threadIdx.x;
    if (i < n) p[i] = 0.0f;
}

// fp32 -> packed bf16 pairs (optionally fused ReLU)
__global__ void cvt_bf16_pack(const float* __restrict__ x, unsigned* __restrict__ out,
                              int npairs, int relu) {
    int i = blockIdx.x * blockDim.x + threadIdx.x;
    if (i >= npairs) return;
    float2 v = ((const float2*)x)[i];
    if (relu) { v.x = fmaxf(v.x, 0.0f); v.y = fmaxf(v.y, 0.0f); }
    out[i] = pack_bf16(v.x, v.y);
}

// Weight [K][N] fp32 row-major -> packed pair-major [K/2][N] u32
__global__ void pack_weight(const float* __restrict__ W, unsigned* __restrict__ out,
                            int K, int N) {
    int i = blockIdx.x * blockDim.x + threadIdx.x;
    int total = (K >> 1) * N;
    if (i >= total) return;
    int kp = i / N, n = i - kp * N;
    out[i] = pack_bf16(W[(2 * kp) * N + n], W[(2 * kp + 1) * N + n]);
}

__global__ void deg_count(const int* __restrict__ dst, float* __restrict__ deg, int E) {
    int e = blockIdx.x * blockDim.x + threadIdx.x;
    if (e < E) unsafeAtomicAdd(&deg[dst[e]], 1.0f);
}

// dis[i] = rsqrt(deg[i] + 1)   (+1 = self loop), in place
__global__ void deg_finalize(float* d, int n) {
    int i = blockIdx.x * blockDim.x + threadIdx.x;
    if (i < n) d[i] = rsqrtf(d[i] + 1.0f);
}

// ---------------------------------------------------------------------------
// bf16 WMMA GEMM:  C[M,N] = A[M,K] * B[K,N],  K == 256
// A: bf16 row-major, viewed as u32 pairs (K/2 per row)
// Bp: packed pair-major [K/2][N] u32
// Block = 256 threads (8 waves), tile = 128 rows x 64 cols
// ---------------------------------------------------------------------------
#define GK 256
__global__ __launch_bounds__(256)
void gemm_bf16(const unsigned* __restrict__ A, const unsigned* __restrict__ Bp,
               float* __restrict__ C, int M, int N) {
    __shared__ unsigned Atile[128 * 20];   // [row][kpair], stride 20 u32 (16B aligned)
    __shared__ unsigned Btile[64 * 20];    // transposed: [col][kpair]

    const int rowBase = blockIdx.x * 128;
    const int colBase = blockIdx.y * 64;
    const int tid  = threadIdx.x;
    const int wave = tid >> 5;
    const int lane = tid & 31;
    const int half = lane >> 4;
    const int mn   = lane & 15;
    const int Ku   = GK >> 1;              // u32 per A row = 128

    v8f acc[4];
#pragma unroll
    for (int c = 0; c < 4; ++c)
#pragma unroll
        for (int j = 0; j < 8; ++j) acc[c][j] = 0.0f;

    for (int kt = 0; kt < GK / 32; ++kt) {
        // ---- cooperative A tile load: 128 rows x 16 u32, as 512 x b128 ----
#pragma unroll
        for (int rep = 0; rep < 2; ++rep) {
            int idx4 = tid + rep * 256;    // uint4 index 0..511
            int r  = idx4 >> 2;
            int cq = idx4 & 3;
            U4 val = {0u, 0u, 0u, 0u};
            int grow = rowBase + r;
            if (grow < M)
                val = *(const U4*)&A[(size_t)grow * Ku + kt * 16 + cq * 4];
            *(U4*)&Atile[r * 20 + cq * 4] = val;
        }
        // ---- cooperative B tile load (transpose into [col][kpair]) ----
#pragma unroll
        for (int rep = 0; rep < 4; ++rep) {
            int idx = tid + rep * 256;     // 0..1023
            int kp  = idx >> 6;            // 0..15
            int n   = idx & 63;
            Btile[n * 20 + kp] = Bp[(size_t)(kt * 16 + kp) * N + colBase + n];
        }
        __syncthreads();

        // ---- A fragment (16x32 bf16): kpair(v) = {v + 4*half, 8+(v-4)+4*half} ----
        v16bf av;
        ((U4*)&av)[0] = *(const U4*)&Atile[(wave * 16 + mn) * 20 + 4 * half];
        ((U4*)&av)[1] = *(const U4*)&Atile[(wave * 16 + mn) * 20 + 8 + 4 * half];

#pragma unroll
        for (int c = 0; c < 4; ++c) {
            // B fragment (32x16 bf16): kpair(v) = {v + 4*half, 8+(v-4)+4*half}
            v16bf bv;
            ((U4*)&bv)[0] = *(const U4*)&Btile[(c * 16 + mn) * 20 + 4 * half];
            ((U4*)&bv)[1] = *(const U4*)&Btile[(c * 16 + mn) * 20 + 8 + 4 * half];
            acc[c] = __builtin_amdgcn_wmma_f32_16x16x32_bf16(
                false, av, false, bv, (short)0, acc[c], false, false);
        }
        __syncthreads();
    }

    // ---- store: C/D layout row = i + 8*half, col = lane%16 ----
#pragma unroll
    for (int c = 0; c < 4; ++c) {
        int col = colBase + c * 16 + mn;
#pragma unroll
        for (int i = 0; i < 8; ++i) {
            int row = rowBase + wave * 16 + i + 8 * half;
            if (row < M) C[(size_t)row * N + col] = acc[c][i];
        }
    }
}

// ---------------------------------------------------------------------------
// Self-loop + bias init:  out = xwA*disA^2 + bA [+ xwB*disB^2 + bB]
// ---------------------------------------------------------------------------
__global__ void init_two(float* __restrict__ out,
                         const float* __restrict__ xwA, const float* __restrict__ xwB,
                         const float* __restrict__ disA, const float* __restrict__ disB,
                         const float* __restrict__ bA, const float* __restrict__ bB,
                         int M, int H) {
    int idx = blockIdx.x * blockDim.x + threadIdx.x;
    if (idx >= M * H) return;
    int i = idx / H, f = idx - i * H;
    float da = disA[i], db = disB[i];
    out[idx] = xwA[idx] * da * da + bA[f] + xwB[idx] * db * db + bB[f];
}

__global__ void init_one(float* __restrict__ out, const float* __restrict__ xw,
                         const float* __restrict__ dis, const float* __restrict__ b,
                         int M, int H) {
    int idx = blockIdx.x * blockDim.x + threadIdx.x;
    if (idx >= M * H) return;
    int i = idx / H, f = idx - i * H;
    float d = dis[i];
    out[idx] = xw[idx] * d * d + b[f];
}

// ---------------------------------------------------------------------------
// Edge scatter: out[dst] += xw[src] * dis[src]*dis[dst]   (float4 per thread)
// ---------------------------------------------------------------------------
__global__ void scatter_edges(const float* __restrict__ xw,
                              const int* __restrict__ src, const int* __restrict__ dst,
                              const float* __restrict__ dis, float* __restrict__ out,
                              int E, int H) {
    int chunks = H >> 2;
    int idx = blockIdx.x * blockDim.x + threadIdx.x;
    if (idx >= E * chunks) return;
    int e = idx / chunks, c = idx - e * chunks;
    int s = src[e], d = dst[e];
    float w = dis[s] * dis[d];
    const float4 v = *(const float4*)&xw[(size_t)s * H + c * 4];
    float* o = &out[(size_t)d * H + c * 4];
    unsafeAtomicAdd(o + 0, v.x * w);
    unsafeAtomicAdd(o + 1, v.y * w);
    unsafeAtomicAdd(o + 2, v.z * w);
    unsafeAtomicAdd(o + 3, v.w * w);
}

// ---------------------------------------------------------------------------
// Link prediction: pred[e] = dot(g2[src[e]], c2[dst[e]]), H2 = 128
// One 32-lane group per edge; wave32 shuffle reduction.
// ---------------------------------------------------------------------------
__global__ __launch_bounds__(256)
void edge_dot(const float* __restrict__ g, const float* __restrict__ c,
              const int* __restrict__ es, const int* __restrict__ ed,
              float* __restrict__ pred, int E) {
    int e = blockIdx.x * 8 + (threadIdx.x >> 5);
    int lane = threadIdx.x & 31;
    if (e >= E) return;
    const float4 a = *(const float4*)&g[(size_t)es[e] * HH2 + lane * 4];
    const float4 b = *(const float4*)&c[(size_t)ed[e] * HH2 + lane * 4];
    float s = a.x * b.x + a.y * b.y + a.z * b.z + a.w * b.w;
#pragma unroll
    for (int m = 16; m >= 1; m >>= 1) s += __shfl_xor(s, m, 32);
    if (lane == 0) pred[e] = s;
}

// ---------------------------------------------------------------------------
// Host orchestration
// ---------------------------------------------------------------------------
static inline int cdiv(int a, int b) { return (a + b - 1) / b; }

extern "C" void kernel_launch(void* const* d_in, const int* in_sizes, int n_in,
                              void* d_out, int out_size, void* d_ws, size_t ws_size,
                              hipStream_t stream) {
    (void)in_sizes; (void)n_in; (void)out_size; (void)ws_size;

    const float* x_gene = (const float*)d_in[0];
    const float* x_cell = (const float*)d_in[1];
    const int* ei_gg  = (const int*)d_in[2];
    const int* ei_rev = (const int*)d_in[3];
    const int* ei_cc  = (const int*)d_in[4];
    const int* eli_src = (const int*)d_in[5];
    const int* eli_dst = (const int*)d_in[6];
    const float* W1_gg  = (const float*)d_in[7];  const float* b1_gg  = (const float*)d_in[8];
    const float* W1_rev = (const float*)d_in[9];  const float* b1_rev = (const float*)d_in[10];
    const float* W1_cc  = (const float*)d_in[11]; const float* b1_cc  = (const float*)d_in[12];
    const float* W2_gg  = (const float*)d_in[13]; const float* b2_gg  = (const float*)d_in[14];
    const float* W2_rev = (const float*)d_in[15]; const float* b2_rev = (const float*)d_in[16];
    const float* W2_cc  = (const float*)d_in[17]; const float* b2_cc  = (const float*)d_in[18];
    float* pred = (float*)d_out;

    // edge_index is [2, E]: row 0 = src, row 1 = dst
    const int *src_gg = ei_gg,  *dst_gg = ei_gg + EGG;
    const int *src_rv = ei_rev, *dst_rv = ei_rev + EGG;
    const int *src_cc = ei_cc,  *dst_cc = ei_cc + ECC;

    // ---- workspace carve ----
    char* ws = (char*)d_ws;
    size_t off = 0;
    auto carve = [&](size_t bytes) -> char* {
        char* p = ws + off;
        off += (bytes + 255) & ~(size_t)255;
        return p;
    };
    unsigned* XGH   = (unsigned*)carve((size_t)NG * FIN * 2);  // x_gene bf16 packed
    unsigned* XCH   = (unsigned*)carve((size_t)NC * FIN * 2);
    unsigned* W1ggP = (unsigned*)carve((size_t)(FIN / 2) * HH1 * 4);
    unsigned* W1rvP = (unsigned*)carve((size_t)(FIN / 2) * HH1 * 4);
    unsigned* W1ccP = (unsigned*)carve((size_t)(FIN / 2) * HH1 * 4);
    unsigned* W2ggP = (unsigned*)carve((size_t)(HH1 / 2) * HH2 * 4);
    unsigned* W2rvP = (unsigned*)carve((size_t)(HH1 / 2) * HH2 * 4);
    unsigned* W2ccP = (unsigned*)carve((size_t)(HH1 / 2) * HH2 * 4);
    float* DGG = (float*)carve((size_t)NG * 4);                // deg -> dis in place
    float* DRV = (float*)carve((size_t)NG * 4);
    float* DCC = (float*)carve((size_t)NC * 4);
    float* XW_A = (float*)carve((size_t)NG * HH1 * 4);         // xw1_gg, later xw2_gg
    float* XW_B = (float*)carve((size_t)NG * HH1 * 4);         // xw1_rev, later xw2_rev
    float* XW_C = (float*)carve((size_t)NC * HH1 * 4);         // xw1_cc, later xw2_cc
    float* G1  = (float*)carve((size_t)NG * HH1 * 4);
    float* C1  = (float*)carve((size_t)NC * HH1 * 4);
    unsigned* G1H = (unsigned*)carve((size_t)NG * HH1 * 2);
    unsigned* C1H = (unsigned*)carve((size_t)NC * HH1 * 2);
    float* G2 = (float*)carve((size_t)NG * HH2 * 4);
    float* C2 = (float*)carve((size_t)NC * HH2 * 4);

    const int B = 256;

    // ---- degrees (per relation, dst-count + self loop) ----
    zero_f32<<<cdiv(NG, B), B, 0, stream>>>(DGG, NG);
    zero_f32<<<cdiv(NG, B), B, 0, stream>>>(DRV, NG);
    zero_f32<<<cdiv(NC, B), B, 0, stream>>>(DCC, NC);
    deg_count<<<cdiv(EGG, B), B, 0, stream>>>(dst_gg, DGG, EGG);
    deg_count<<<cdiv(EGG, B), B, 0, stream>>>(dst_rv, DRV, EGG);
    deg_count<<<cdiv(ECC, B), B, 0, stream>>>(dst_cc, DCC, ECC);
    deg_finalize<<<cdiv(NG, B), B, 0, stream>>>(DGG, NG);
    deg_finalize<<<cdiv(NG, B), B, 0, stream>>>(DRV, NG);
    deg_finalize<<<cdiv(NC, B), B, 0, stream>>>(DCC, NC);

    // ---- bf16 conversions ----
    cvt_bf16_pack<<<cdiv(NG * FIN / 2, B), B, 0, stream>>>(x_gene, XGH, NG * FIN / 2, 0);
    cvt_bf16_pack<<<cdiv(NC * FIN / 2, B), B, 0, stream>>>(x_cell, XCH, NC * FIN / 2, 0);
    pack_weight<<<cdiv(FIN / 2 * HH1, B), B, 0, stream>>>(W1_gg,  W1ggP, FIN, HH1);
    pack_weight<<<cdiv(FIN / 2 * HH1, B), B, 0, stream>>>(W1_rev, W1rvP, FIN, HH1);
    pack_weight<<<cdiv(FIN / 2 * HH1, B), B, 0, stream>>>(W1_cc,  W1ccP, FIN, HH1);
    pack_weight<<<cdiv(HH1 / 2 * HH2, B), B, 0, stream>>>(W2_gg,  W2ggP, HH1, HH2);
    pack_weight<<<cdiv(HH1 / 2 * HH2, B), B, 0, stream>>>(W2_rev, W2rvP, HH1, HH2);
    pack_weight<<<cdiv(HH1 / 2 * HH2, B), B, 0, stream>>>(W2_cc,  W2ccP, HH1, HH2);

    // ---- layer 1 GEMMs (WMMA) ----
    gemm_bf16<<<dim3(cdiv(NG, 128), HH1 / 64), B, 0, stream>>>(XGH, W1ggP, XW_A, NG, HH1);
    gemm_bf16<<<dim3(cdiv(NG, 128), HH1 / 64), B, 0, stream>>>(XGH, W1rvP, XW_B, NG, HH1);
    gemm_bf16<<<dim3(cdiv(NC, 128), HH1 / 64), B, 0, stream>>>(XCH, W1ccP, XW_C, NC, HH1);

    // ---- layer 1 aggregation ----
    init_two<<<cdiv(NG * HH1, B), B, 0, stream>>>(G1, XW_A, XW_B, DGG, DRV, b1_gg, b1_rev, NG, HH1);
    init_one<<<cdiv(NC * HH1, B), B, 0, stream>>>(C1, XW_C, DCC, b1_cc, NC, HH1);
    scatter_edges<<<cdiv(EGG * (HH1 / 4), B), B, 0, stream>>>(XW_A, src_gg, dst_gg, DGG, G1, EGG, HH1);
    scatter_edges<<<cdiv(EGG * (HH1 / 4), B), B, 0, stream>>>(XW_B, src_rv, dst_rv, DRV, G1, EGG, HH1);
    scatter_edges<<<cdiv(ECC * (HH1 / 4), B), B, 0, stream>>>(XW_C, src_cc, dst_cc, DCC, C1, ECC, HH1);

    // ---- ReLU + bf16 for layer 2 ----
    cvt_bf16_pack<<<cdiv(NG * HH1 / 2, B), B, 0, stream>>>(G1, G1H, NG * HH1 / 2, 1);
    cvt_bf16_pack<<<cdiv(NC * HH1 / 2, B), B, 0, stream>>>(C1, C1H, NC * HH1 / 2, 1);

    // ---- layer 2 GEMMs (WMMA), reuse XW buffers ----
    gemm_bf16<<<dim3(cdiv(NG, 128), HH2 / 64), B, 0, stream>>>(G1H, W2ggP, XW_A, NG, HH2);
    gemm_bf16<<<dim3(cdiv(NG, 128), HH2 / 64), B, 0, stream>>>(G1H, W2rvP, XW_B, NG, HH2);
    gemm_bf16<<<dim3(cdiv(NC, 128), HH2 / 64), B, 0, stream>>>(C1H, W2ccP, XW_C, NC, HH2);

    // ---- layer 2 aggregation ----
    init_two<<<cdiv(NG * HH2, B), B, 0, stream>>>(G2, XW_A, XW_B, DGG, DRV, b2_gg, b2_rev, NG, HH2);
    init_one<<<cdiv(NC * HH2, B), B, 0, stream>>>(C2, XW_C, DCC, b2_cc, NC, HH2);
    scatter_edges<<<cdiv(EGG * (HH2 / 4), B), B, 0, stream>>>(XW_A, src_gg, dst_gg, DGG, G2, EGG, HH2);
    scatter_edges<<<cdiv(EGG * (HH2 / 4), B), B, 0, stream>>>(XW_B, src_rv, dst_rv, DRV, G2, EGG, HH2);
    scatter_edges<<<cdiv(ECC * (HH2 / 4), B), B, 0, stream>>>(XW_C, src_cc, dst_cc, DCC, C2, ECC, HH2);

    // ---- link prediction ----
    edge_dot<<<cdiv(ELBL, 8), B, 0, stream>>>(G2, C2, eli_src, eli_dst, pred, ELBL);
}
